// PerceiverAttention_79018808312129
// MI455X (gfx1250) — matrix operand order repaired and checked by reading, not compiled, not run on hardware
//
#include <hip/hip_runtime.h>
#include <hip/hip_bf16.h>
#include <stdint.h>

typedef __bf16 bf16_t;
typedef __attribute__((ext_vector_type(16))) __bf16 v16bf;
typedef __attribute__((ext_vector_type(8)))  float   v8f;
typedef __attribute__((ext_vector_type(4)))  unsigned int u32x4;

namespace {
constexpr int kNX = 256, kNZ = 64, kD = 1024;
constexpr int kHeads = 8, kDh = 64, kInner = 512;
constexpr int kG  = 8 * 16;          // B*T = 128 groups
constexpr int kRC = kNX + kNZ;       // 320 rows of concat(xn, zn)
constexpr int kLda = 32 + 8;         // LDS row stride (bf16) for 32-wide K tiles
}

union FragU { v16bf v; u32x4 h[2]; };

// ---- CDNA5 async global->LDS copy (ASYNCcnt-tracked, bypasses VGPRs) -------
__device__ __forceinline__ void async_cpy16(void* ldsPtr, const void* gPtr) {
  // VDST = per-lane LDS byte address (flat shared addr truncates to LDS offset),
  // VADDR = per-lane 64-bit global address.
  asm volatile("global_load_async_to_lds_b128 %0, %1, off"
               :: "v"((unsigned)(uintptr_t)ldsPtr),
                  "v"((unsigned long long)(uintptr_t)gPtr)
               : "memory");
}
__device__ __forceinline__ void wait_async0() {
#if __has_builtin(__builtin_amdgcn_s_wait_asynccnt)
  __builtin_amdgcn_s_wait_asynccnt(0);
#else
  asm volatile("s_wait_asynccnt 0x0" ::: "memory");
#endif
}

__device__ __forceinline__ v8f v8f_zero() {
  v8f z;
#pragma unroll
  for (int i = 0; i < 8; ++i) z[i] = 0.f;
  return z;
}

__device__ __forceinline__ v8f wmma_bf16(v16bf a, v16bf b, v8f c) {
  // D = A(16x32 bf16) x B(32x16 bf16) + C(f32)
  return __builtin_amdgcn_wmma_f32_16x16x32_bf16(false, a, false, b, (short)0, c,
                                                 false, false);
}

// 16-lane-half butterfly reductions (C-matrix columns live in 16-lane halves)
__device__ __forceinline__ float half16_max(float v) {
#pragma unroll
  for (int m = 8; m >= 1; m >>= 1) v = fmaxf(v, __shfl_xor(v, m, 32));
  return v;
}
__device__ __forceinline__ float half16_sum(float v) {
#pragma unroll
  for (int m = 8; m >= 1; m >>= 1) v += __shfl_xor(v, m, 32);
  return v;
}

// ---------------------------------------------------------------------------
// Weight convert + transpose: Wt[n*K + k] = bf16(W[k*N + n])
// ---------------------------------------------------------------------------
__global__ void wconv_kernel(const float* __restrict__ W, bf16_t* __restrict__ Wt,
                             int K, int N) {
  long idx = (long)blockIdx.x * 256 + threadIdx.x;
  if (idx < (long)N * K) {
    int n = (int)(idx / K), k = (int)(idx % K);
    Wt[idx] = (bf16_t)W[(long)k * N + n];
  }
}

// ---------------------------------------------------------------------------
// LayerNorm (x rows 0..255, latent rows 256..319 per group) -> bf16 concat
// one block (256 threads) per row of 1024
// ---------------------------------------------------------------------------
__global__ __launch_bounds__(256) void ln_kernel(
    const float* __restrict__ x, const float* __restrict__ lat,
    const float* __restrict__ gm, const float* __restrict__ bm,
    const float* __restrict__ gl, const float* __restrict__ bl,
    bf16_t* __restrict__ cat) {
  int row = blockIdx.x;           // 0 .. kG*kRC-1
  int g = row / kRC, r = row % kRC;
  const float *src, *gg, *bb;
  if (r < kNX) { src = x   + ((long)g * kNX + r)        * kD; gg = gm; bb = bm; }
  else         { src = lat + ((long)g * kNZ + (r - kNX)) * kD; gg = gl; bb = bl; }

  int tid = threadIdx.x, lane = tid & 31, w = tid >> 5;
  float4 f = ((const float4*)src)[tid];   // 256 threads * 4 = 1024
  float s  = f.x + f.y + f.z + f.w;
  float s2 = f.x * f.x + f.y * f.y + f.z * f.z + f.w * f.w;
#pragma unroll
  for (int m = 16; m >= 1; m >>= 1) {
    s  += __shfl_xor(s,  m, 32);
    s2 += __shfl_xor(s2, m, 32);
  }
  __shared__ float red[16];
  if (lane == 0) { red[w] = s; red[8 + w] = s2; }
  __syncthreads();
  float ts = 0.f, ts2 = 0.f;
#pragma unroll
  for (int i = 0; i < 8; ++i) { ts += red[i]; ts2 += red[8 + i]; }
  const float mu  = ts * (1.f / 1024.f);
  const float var = ts2 * (1.f / 1024.f) - mu * mu;
  const float inv = rsqrtf(var + 1e-5f);

  bf16_t* dst = cat + (long)row * kD + tid * 4;
  int c0 = tid * 4;
  float vv[4] = {f.x, f.y, f.z, f.w};
#pragma unroll
  for (int e = 0; e < 4; ++e)
    dst[e] = (bf16_t)((vv[e] - mu) * inv * gg[c0 + e] + bb[c0 + e]);
}

// ---------------------------------------------------------------------------
// Templated bf16 WMMA GEMM: C(MxN) = A(MxK, row-major) x Bt(NxK, "[n][k]")
// 8 waves, wave tile 32x64 (2x4 tiles of 16x16), BK=32.
// Double-buffered LDS fed by GLOBAL_LOAD_ASYNC_TO_LDS_B128 (ASYNCcnt pipeline).
// EPI: 0 = kv split (k row-major / v transposed), 1 = q bf16, 2 = f32 out
// ---------------------------------------------------------------------------
template <int BM, int BN, int EPI>
__global__ __launch_bounds__(256) void gemm_kernel(
    const bf16_t* __restrict__ Abase, long long aStride,
    const bf16_t* __restrict__ Bt, int M, int K, int N,
    bf16_t* __restrict__ kOut, bf16_t* __restrict__ vtOut,
    bf16_t* __restrict__ qOut, float* __restrict__ fOut) {
  constexpr int WCOLS = BN / 64;
  static_assert((BM / 32) * (BN / 64) == 8, "8 waves");
  __shared__ alignas(16) bf16_t lsA[2][BM * kLda];
  __shared__ alignas(16) bf16_t lsB[2][BN * kLda];

  const int tid = threadIdx.x, lane = tid & 31, ln = lane & 15, lh = lane >> 4;
  const int w = tid >> 5, wr = w / WCOLS, wc = w % WCOLS;
  const int g = blockIdx.z;
  const int m0 = blockIdx.x * BM, n0 = blockIdx.y * BN;
  const bf16_t* A = Abase + (long long)g * aStride;

  // Issue this thread's share of one K-slice as async copies into buffer `buf`.
  auto stage = [&](int buf, int ks) {
#pragma unroll
    for (int c = tid; c < BM * 4; c += 256) {
      int row = c >> 2, part = c & 3;
      int gr = m0 + row; if (gr > M - 1) gr = M - 1;   // clamp tail rows
      async_cpy16(lsA[buf] + row * kLda + part * 8,
                  A + (long)gr * K + ks + part * 8);
    }
#pragma unroll
    for (int c = tid; c < BN * 4; c += 256) {
      int row = c >> 2, part = c & 3;
      async_cpy16(lsB[buf] + row * kLda + part * 8,
                  Bt + (long)(n0 + row) * K + ks + part * 8);
    }
  };

  v8f acc[2][4];
#pragma unroll
  for (int i = 0; i < 2; ++i)
#pragma unroll
    for (int j = 0; j < 4; ++j) acc[i][j] = v8f_zero();

  stage(0, 0);
  wait_async0();
  __syncthreads();

  int buf = 0;
  for (int ks = 0; ks < K; ks += 32) {
    if (ks + 32 < K) stage(buf ^ 1, ks + 32);   // prefetch next slice (async)

    FragU a[2], b[4];
#pragma unroll
    for (int ti = 0; ti < 2; ++ti) {
      int r = wr * 32 + ti * 16 + ln;
      a[ti].h[0] = *(const u32x4*)(lsA[buf] + r * kLda + 8 * lh);      // k=8h..
      a[ti].h[1] = *(const u32x4*)(lsA[buf] + r * kLda + 16 + 8 * lh); // k=16+8h..
    }
#pragma unroll
    for (int tj = 0; tj < 4; ++tj) {
      int r = wc * 64 + tj * 16 + ln;
      b[tj].h[0] = *(const u32x4*)(lsB[buf] + r * kLda + 8 * lh);      // k=8h..
      b[tj].h[1] = *(const u32x4*)(lsB[buf] + r * kLda + 16 + 8 * lh);
    }
#pragma unroll
    for (int ti = 0; ti < 2; ++ti)
#pragma unroll
      for (int tj = 0; tj < 4; ++tj)
        acc[ti][tj] = wmma_bf16(a[ti].v, b[tj].v, acc[ti][tj]);

    wait_async0();      // next slice landed (this wave's share)
    __syncthreads();    // everyone done reading buf & everyone's copies done
    buf ^= 1;
  }

#pragma unroll
  for (int ti = 0; ti < 2; ++ti)
#pragma unroll
    for (int tj = 0; tj < 4; ++tj)
#pragma unroll
      for (int r = 0; r < 8; ++r) {
        int m = m0 + wr * 32 + ti * 16 + lh * 8 + r;
        int n = n0 + wc * 64 + tj * 16 + ln;
        float val = acc[ti][tj][r];
        if constexpr (EPI == 0) {        // kv: k row-major, v transposed
          if (m < M) {
            if (n < kInner) {
              int h = n >> 6, d = n & 63;
              kOut[(((long)g * kHeads + h) * kRC + m) * kDh + d] = (bf16_t)val;
            } else {
              int c = n - kInner, h = c >> 6, d = c & 63;
              vtOut[(((long)g * kHeads + h) * kDh + d) * kRC + m] = (bf16_t)val;
            }
          }
        } else if constexpr (EPI == 1) { // q -> (g,h,i,d)
          int h = n >> 6, d = n & 63;
          qOut[(((long)g * kHeads + h) * kNZ + m) * kDh + d] = (bf16_t)val;
        } else {                         // final fp32 output (g, i, D)
          fOut[((long)g * kNZ + m) * kD + n] = val;
        }
      }
}

// ---------------------------------------------------------------------------
// Flash attention per (head, group): 4 waves x 16 query rows, j-steps of 32
// K/V blocks staged to LDS with async copies.
// ---------------------------------------------------------------------------
__global__ __launch_bounds__(128) void attn_kernel(
    const bf16_t* __restrict__ q, const bf16_t* __restrict__ kmat,
    const bf16_t* __restrict__ vt, const unsigned char* __restrict__ mask,
    bf16_t* __restrict__ attnout) {
  const int h = blockIdx.x, g = blockIdx.y;
  const int tid = threadIdx.x, lane = tid & 31, ln = lane & 15, lh = lane >> 4;
  const int w = tid >> 5;

  if (!mask[g]) {  // whole group masked -> zero attn output slab for this head
    for (int idx = tid; idx < kNZ * kDh; idx += 128) {
      int i = idx >> 6, d = idx & 63;
      attnout[((long)g * kNZ + i) * kInner + h * kDh + d] = (bf16_t)0.f;
    }
    return;
  }

  const bf16_t* qh = q    + (((long)g * kHeads + h) * kNZ) * kDh;
  const bf16_t* kh = kmat + (((long)g * kHeads + h) * kRC) * kDh;
  const bf16_t* vh = vt   + (((long)g * kHeads + h) * kDh) * kRC;

  __shared__ alignas(16) bf16_t lsK[32 * 72];       // [j][d], stride 72
  __shared__ alignas(16) bf16_t lsV[64 * 40];       // [d][j], stride 40
  __shared__ alignas(16) bf16_t lsP[4 * 16 * 40];   // wave-private P tiles

  const int i0 = w * 16;
  FragU qa[2];
#pragma unroll
  for (int kd = 0; kd < 2; ++kd) {
    qa[kd].h[0] = *(const u32x4*)(qh + (i0 + ln) * kDh + kd * 32 + 8 * lh);
    qa[kd].h[1] = *(const u32x4*)(qh + (i0 + ln) * kDh + kd * 32 + 16 + 8 * lh);
  }

  v8f o[4];
#pragma unroll
  for (int t = 0; t < 4; ++t) o[t] = v8f_zero();
  float mrun[8], lrun[8];
#pragma unroll
  for (int r = 0; r < 8; ++r) { mrun[r] = -1e30f; lrun[r] = 0.f; }

  for (int j0 = 0; j0 < kRC; j0 += 32) {
#pragma unroll
    for (int c = tid; c < 32 * 8; c += 128) {    // stage K block (32 x 64)
      int jr = c >> 3, part = c & 7;
      async_cpy16(lsK + jr * 72 + part * 8,
                  kh + (long)(j0 + jr) * kDh + part * 8);
    }
#pragma unroll
    for (int c = tid; c < 64 * 4; c += 128) {    // stage V^T block (64 x 32)
      int dr = c >> 2, part = c & 3;
      async_cpy16(lsV + dr * 40 + part * 8,
                  vh + (long)dr * kRC + j0 + part * 8);
    }
    wait_async0();
    __syncthreads();

    v8f s[2]; s[0] = v8f_zero(); s[1] = v8f_zero();
#pragma unroll
    for (int kd = 0; kd < 2; ++kd)
#pragma unroll
      for (int tj = 0; tj < 2; ++tj) {
        FragU kb;
        int jr = tj * 16 + ln;
        kb.h[0] = *(const u32x4*)(lsK + jr * 72 + kd * 32 + 8 * lh);
        kb.h[1] = *(const u32x4*)(lsK + jr * 72 + kd * 32 + 16 + 8 * lh);
        s[tj] = wmma_bf16(qa[kd].v, kb.v, s[tj]);
      }

    // online softmax; row m = 8*lh + r lives in vgpr r across a 16-lane half
#pragma unroll
    for (int r = 0; r < 8; ++r) {
      float a0 = s[0][r] * 0.125f, a1 = s[1][r] * 0.125f;  // dhead^-0.5
      float tm = half16_max(fmaxf(a0, a1));
      float mnew  = fmaxf(mrun[r], tm);
      float alpha = __expf(mrun[r] - mnew);
      float p0 = __expf(a0 - mnew), p1 = __expf(a1 - mnew);
      float rs = half16_sum(p0 + p1);
      lrun[r] = lrun[r] * alpha + rs;
      mrun[r] = mnew;
#pragma unroll
      for (int t = 0; t < 4; ++t) o[t][r] *= alpha;
      s[0][r] = p0; s[1][r] = p1;
    }

    // C-layout f32 -> A-layout bf16 via wave-private LDS
    bf16_t* pw = lsP + w * 16 * 40;
#pragma unroll
    for (int tj = 0; tj < 2; ++tj)
#pragma unroll
      for (int r = 0; r < 8; ++r)
        pw[(lh * 8 + r) * 40 + tj * 16 + ln] = (bf16_t)s[tj][r];

    FragU pa;
    pa.h[0] = *(const u32x4*)(pw + ln * 40 + 8 * lh);
    pa.h[1] = *(const u32x4*)(pw + ln * 40 + 16 + 8 * lh);

#pragma unroll
    for (int t = 0; t < 4; ++t) {
      FragU vb;
      int dr = t * 16 + ln;
      vb.h[0] = *(const u32x4*)(lsV + dr * 40 + 8 * lh);
      vb.h[1] = *(const u32x4*)(lsV + dr * 40 + 16 + 8 * lh);
      o[t] = wmma_bf16(pa.v, vb.v, o[t]);
    }
    __syncthreads();
  }

#pragma unroll
  for (int r = 0; r < 8; ++r) {
    float inv = 1.0f / lrun[r];
    int i = i0 + lh * 8 + r;
#pragma unroll
    for (int t = 0; t < 4; ++t)
      attnout[((long)g * kNZ + i) * kInner + h * kDh + t * 16 + ln] =
          (bf16_t)(o[t][r] * inv);
  }
}

// ---------------------------------------------------------------------------
extern "C" void kernel_launch(void* const* d_in, const int* in_sizes, int n_in,
                              void* d_out, int out_size, void* d_ws, size_t ws_size,
                              hipStream_t stream) {
  const float* x   = (const float*)d_in[0];
  const float* lat = (const float*)d_in[1];
  const unsigned char* mask = (const unsigned char*)d_in[2];  // jax bool = 1 byte
  const float* gm = (const float*)d_in[3];
  const float* bm = (const float*)d_in[4];
  const float* gl = (const float*)d_in[5];
  const float* bl = (const float*)d_in[6];
  const float* Wq  = (const float*)d_in[7];
  const float* Wkv = (const float*)d_in[8];
  const float* Wo  = (const float*)d_in[9];
  float* out = (float*)d_out;
  (void)in_sizes; (void)n_in; (void)out_size; (void)ws_size;

  char* ws = (char*)d_ws;
  size_t off = 0;
  auto alloc = [&](size_t bytes) {
    void* p = ws + off;
    off = (off + bytes + 255) & ~(size_t)255;
    return p;
  };
  bf16_t* cat  = (bf16_t*)alloc((size_t)kG * kRC * kD * 2);          // 80 MB
  bf16_t* WqT  = (bf16_t*)alloc((size_t)kInner * kD * 2);
  bf16_t* WkvT = (bf16_t*)alloc((size_t)kD * kD * 2);
  bf16_t* WoT  = (bf16_t*)alloc((size_t)kD * kInner * 2);
  bf16_t* qb   = (bf16_t*)alloc((size_t)kG * kHeads * kNZ * kDh * 2);
  bf16_t* kb   = (bf16_t*)alloc((size_t)kG * kHeads * kRC * kDh * 2); // 40 MB
  bf16_t* vtb  = (bf16_t*)alloc((size_t)kG * kHeads * kDh * kRC * 2); // 40 MB
  bf16_t* ao   = (bf16_t*)alloc((size_t)kG * kNZ * kInner * 2);

  // 1) weights -> bf16, [N][K]
  wconv_kernel<<<dim3((kInner * kD + 255) / 256), 256, 0, stream>>>(Wq, WqT, kD, kInner);
  wconv_kernel<<<dim3((kD * kD + 255) / 256), 256, 0, stream>>>(Wkv, WkvT, kD, kD);
  wconv_kernel<<<dim3((kD * kInner + 255) / 256), 256, 0, stream>>>(Wo, WoT, kInner, kD);

  // 2) layernorm x + latents into concat buffer
  ln_kernel<<<dim3(kG * kRC), 256, 0, stream>>>(x, lat, gm, bm, gl, bl, cat);

  // 3) kv = cat @ Wkv  (M=320, K=1024, N=1024), split into k / v^T
  gemm_kernel<128, 128, 0><<<dim3(3, 8, kG), 256, 0, stream>>>(
      cat, (long long)kRC * kD, WkvT, kRC, kD, kD, kb, vtb, nullptr, nullptr);

  // 4) q = zn @ Wq  (M=64, K=1024, N=512); zn = rows 256..319 of cat
  gemm_kernel<64, 256, 1><<<dim3(1, 2, kG), 256, 0, stream>>>(
      cat + (size_t)kNX * kD, (long long)kRC * kD, WqT, kNZ, kD, kInner,
      nullptr, nullptr, qb, nullptr);

  // 5) attention per (head, group)
  attn_kernel<<<dim3(kHeads, kG), 128, 0, stream>>>(qb, kb, vtb, mask, ao);

  // 6) out = attnout @ Wo  (M=64, K=512, N=1024) -> fp32 d_out
  gemm_kernel<64, 256, 2><<<dim3(1, 4, kG), 256, 0, stream>>>(
      ao, (long long)kNZ * kInner, WoT, kNZ, kInner, kD,
      nullptr, nullptr, nullptr, out);
}
